// APPNP_Net_60859686584878
// MI455X (gfx1250) — matrix operand order repaired and checked by reading, not compiled, not run on hardware
//
#include <hip/hip_runtime.h>
#include <hip/hip_bf16.h>

typedef __attribute__((ext_vector_type(16))) _Float16 v16h;
typedef __attribute__((ext_vector_type(8)))  float    v8f;

#define N_NODES   100000
#define N_EDGES   1600000
#define N_FEAT    500
#define HIDDEN    128
#define N_CLASSES 40
#define K_STEPS   10
#define ALPHA     0.1f

#define KPAD1     512     // N_FEAT padded for GEMM1
#define NPAD2     48      // N_CLASSES padded for GEMM2

// ---------------------------------------------------------------------------
// Degree / normalization kernels
// ---------------------------------------------------------------------------
__global__ void deg_init_kernel(float* __restrict__ deg) {
    int i = blockIdx.x * blockDim.x + threadIdx.x;
    if (i < N_NODES) deg[i] = 1.0f;   // self-loop contributes 1 to every node
}

__global__ void deg_count_kernel(const long long* __restrict__ dst,
                                 float* __restrict__ deg) {
    int e = blockIdx.x * blockDim.x + threadIdx.x;
    if (e < N_EDGES) atomicAdd(&deg[(int)dst[e]], 1.0f);
}

__global__ void dinv_kernel(float* __restrict__ deg_dinv) {
    int i = blockIdx.x * blockDim.x + threadIdx.x;
    if (i < N_NODES) deg_dinv[i] = rsqrtf(deg_dinv[i]);  // deg >= 1 always
}

__global__ void norm_kernel(const long long* __restrict__ ei,
                            const float* __restrict__ dinv,
                            float* __restrict__ norm) {
    int e = blockIdx.x * blockDim.x + threadIdx.x;
    if (e < N_EDGES) {
        int s = (int)ei[e];
        int d = (int)ei[N_EDGES + e];
        norm[e] = dinv[s] * dinv[d];
    }
}

// ---------------------------------------------------------------------------
// Weight packing: column-major f16, zero-padded. A WMMA B-fragment for
// (n, k0) is then two 8-half runs at W*T[n][k0 + half*8] and W*T[n][k0 + 16 + half*8].
// ---------------------------------------------------------------------------
__global__ void pack_w1_kernel(const float* __restrict__ W1,
                               _Float16* __restrict__ W1T) {  // [HIDDEN][KPAD1]
    int idx = blockIdx.x * blockDim.x + threadIdx.x;
    if (idx < HIDDEN * KPAD1) {
        int n = idx >> 9;          // / KPAD1
        int k = idx & (KPAD1 - 1);
        W1T[idx] = (k < N_FEAT) ? (_Float16)W1[(size_t)k * HIDDEN + n]
                                : (_Float16)0.0f;
    }
}

__global__ void pack_w2_kernel(const float* __restrict__ W2,
                               _Float16* __restrict__ W2T) {  // [NPAD2][HIDDEN]
    int idx = blockIdx.x * blockDim.x + threadIdx.x;
    if (idx < NPAD2 * HIDDEN) {
        int n = idx >> 7;          // / HIDDEN
        int k = idx & (HIDDEN - 1);
        W2T[idx] = (n < N_CLASSES) ? (_Float16)W2[(size_t)k * N_CLASSES + n]
                                   : (_Float16)0.0f;
    }
}

// ---------------------------------------------------------------------------
// GEMM1: h1 = relu(x @ W1 + b1). One 16-row M-tile per block, 8 waves cover
// the 128 columns. 15 guard-free K-chunks + explicit tail for K=480..499.
// ---------------------------------------------------------------------------
__global__ __launch_bounds__(256) void gemm1_relu_kernel(
    const float* __restrict__ x,      // [N_NODES, N_FEAT]
    const _Float16* __restrict__ W1T, // [HIDDEN, KPAD1] packed
    const float* __restrict__ b1,     // [HIDDEN]
    _Float16* __restrict__ h1)        // [N_NODES, HIDDEN]
{
    const int wave = threadIdx.x >> 5;   // 0..7 -> N tile
    const int lane = threadIdx.x & 31;
    const int half = lane >> 4;
    const int r    = lane & 15;
    const int m0   = blockIdx.x * 16;
    const int n0   = wave * 16;

    const float*    xrow = x   + (size_t)(m0 + r) * N_FEAT;
    const _Float16* wcol = W1T + (size_t)(n0 + r) * KPAD1;

    v8f c = {};

    // 15 full K-chunks: completely guard-free, all loads 16B-aligned
    for (int k0 = 0; k0 < 480; k0 += 32) {
        v16h a, b;
        const float* p = xrow + k0 + half * 8;      // runs [0..7] and [16..23]
#pragma unroll
        for (int j = 0; j < 8; ++j) {
            a[j]     = (_Float16)p[j];
            a[j + 8] = (_Float16)p[j + 16];
        }
        const _Float16* q = wcol + k0 + half * 8;   // runs [0..7] and [16..23]
#pragma unroll
        for (int j = 0; j < 8; ++j) {
            b[j]     = q[j];
            b[j + 8] = q[j + 16];
        }
        c = __builtin_amdgcn_wmma_f32_16x16x32_f16(false, a, false, b,
                                                   (short)0, c, false, false);
    }

    // Tail chunk k0 = 480 (valid K: 480..499); branch-free via selects
    {
        v16h a, b;
        const float* p = xrow + 480 + half * 8;     // K 480..487 / 488..495
#pragma unroll
        for (int j = 0; j < 8; ++j) a[j] = (_Float16)p[j];
        // second run: half==0 needs K 496..499, half==1 needs K 504..511 (=0)
        const float* t = xrow + 492;                // in-bounds, 16B-aligned
        float t4 = t[4], t5 = t[5], t6 = t[6], t7 = t[7];
        a[8]  = (half == 0) ? (_Float16)t4 : (_Float16)0.0f;
        a[9]  = (half == 0) ? (_Float16)t5 : (_Float16)0.0f;
        a[10] = (half == 0) ? (_Float16)t6 : (_Float16)0.0f;
        a[11] = (half == 0) ? (_Float16)t7 : (_Float16)0.0f;
        a[12] = a[13] = a[14] = a[15] = (_Float16)0.0f;
        const _Float16* q = wcol + 480 + half * 8;  // zero-padded pack
#pragma unroll
        for (int j = 0; j < 8; ++j) {
            b[j]     = q[j];
            b[j + 8] = q[j + 16];
        }
        c = __builtin_amdgcn_wmma_f32_16x16x32_f16(false, a, false, b,
                                                   (short)0, c, false, false);
    }

#pragma unroll
    for (int v = 0; v < 8; ++v) {
        int row = m0 + half * 8 + v;
        int col = n0 + r;
        float val = c[v] + b1[col];
        val = val > 0.0f ? val : 0.0f;
        h1[(size_t)row * HIDDEN + col] = (_Float16)val;
    }
}

// ---------------------------------------------------------------------------
// GEMM2: h0 = h1 @ W2 + b2. N=40 padded to 48 (3 waves per block); padding
// lives in the zero-packed W2T, so the K loop is guard-free.
// ---------------------------------------------------------------------------
__global__ __launch_bounds__(96) void gemm2_kernel(
    const _Float16* __restrict__ h1,  // [N_NODES, HIDDEN]
    const _Float16* __restrict__ W2T, // [NPAD2, HIDDEN] packed
    const float* __restrict__ b2,     // [N_CLASSES]
    float* __restrict__ h0)           // [N_NODES, N_CLASSES]
{
    const int wave = threadIdx.x >> 5;   // 0..2 -> N tile
    const int lane = threadIdx.x & 31;
    const int half = lane >> 4;
    const int r    = lane & 15;
    const int m0   = blockIdx.x * 16;
    const int n    = wave * 16 + r;

    const _Float16* hrow = h1  + (size_t)(m0 + r) * HIDDEN;
    const _Float16* wcol = W2T + (size_t)n * HIDDEN;

    v8f c = {};
#pragma unroll
    for (int k0 = 0; k0 < HIDDEN; k0 += 32) {
        v16h a, b;
        const _Float16* p = hrow + k0 + half * 8;
#pragma unroll
        for (int j = 0; j < 8; ++j) {
            a[j]     = p[j];
            a[j + 8] = p[j + 16];
        }
        const _Float16* q = wcol + k0 + half * 8;
#pragma unroll
        for (int j = 0; j < 8; ++j) {
            b[j]     = q[j];
            b[j + 8] = q[j + 16];
        }
        c = __builtin_amdgcn_wmma_f32_16x16x32_f16(false, a, false, b,
                                                   (short)0, c, false, false);
    }

    if (n < N_CLASSES) {
#pragma unroll
        for (int v = 0; v < 8; ++v) {
            int row = m0 + half * 8 + v;
            h0[(size_t)row * N_CLASSES + n] = c[v] + b2[n];
        }
    }
}

// ---------------------------------------------------------------------------
// Propagation: hnext = alpha*h0 + (1-alpha)*(dinv[i]^2 * hcur[i]  (self loop)
//                                            + sum_e norm[e]*hcur[src])
// ---------------------------------------------------------------------------
__global__ void prop_init_kernel(const float* __restrict__ h0,
                                 const float* __restrict__ hcur,
                                 const float* __restrict__ dinv,
                                 float* __restrict__ hnext) {
    unsigned idx = blockIdx.x * blockDim.x + threadIdx.x;
    if (idx < (unsigned)(N_NODES * N_CLASSES)) {
        int node = idx / N_CLASSES;
        float di = dinv[node];
        hnext[idx] = ALPHA * h0[idx] + (1.0f - ALPHA) * di * di * hcur[idx];
    }
}

__global__ void prop_edge_kernel(const long long* __restrict__ ei,
                                 const float* __restrict__ norm,
                                 const float* __restrict__ hcur,
                                 float* __restrict__ hnext) {
    // one thread per (edge, 4-channel group); 40 channels -> 10 groups
    unsigned t = blockIdx.x * blockDim.x + threadIdx.x;
    const unsigned total = (unsigned)N_EDGES * 10u;
    if (t >= total) return;
    unsigned e = t / 10u;
    unsigned g = (t - e * 10u) * 4u;
    int s = (int)ei[e];
    int d = (int)ei[N_EDGES + e];
    float w = (1.0f - ALPHA) * norm[e];
    const float* hs = hcur + (size_t)s * N_CLASSES + g;
    float* hd = hnext + (size_t)d * N_CLASSES + g;
    atomicAdd(hd + 0, w * hs[0]);
    atomicAdd(hd + 1, w * hs[1]);
    atomicAdd(hd + 2, w * hs[2]);
    atomicAdd(hd + 3, w * hs[3]);
}

// ---------------------------------------------------------------------------
extern "C" void kernel_launch(void* const* d_in, const int* in_sizes, int n_in,
                              void* d_out, int out_size, void* d_ws, size_t ws_size,
                              hipStream_t stream) {
    const float*     x  = (const float*)d_in[0];
    const long long* ei = (const long long*)d_in[1];   // int64 [2, N_EDGES]
    const float*     W1 = (const float*)d_in[2];
    const float*     b1 = (const float*)d_in[3];
    const float*     W2 = (const float*)d_in[4];
    const float*     b2 = (const float*)d_in[5];
    float* out = (float*)d_out;

    // Workspace layout (256B-aligned chunks)
    char* ws = (char*)d_ws;
    float*    dinv = (float*)ws;    ws += 400384;     // 100000*4 padded
    float*    norm = (float*)ws;    ws += 6400000;    // 1.6M*4
    _Float16* W1T  = (_Float16*)ws; ws += HIDDEN * KPAD1 * 2;    // 131072
    _Float16* W2T  = (_Float16*)ws; ws += NPAD2 * HIDDEN * 2;    // 12288
    _Float16* h1   = (_Float16*)ws; ws += 25600000;   // 100000*128*2
    float*    h0   = (float*)ws;    ws += 16000000;   // 100000*40*4
    float*    hA   = (float*)ws;    ws += 16000000;
    float*    hB   = (float*)ws;    ws += 16000000;

    // 1. degrees (in-degree over dst + self loop) and edge norms
    deg_init_kernel<<<(N_NODES + 255) / 256, 256, 0, stream>>>(dinv);
    deg_count_kernel<<<(N_EDGES + 255) / 256, 256, 0, stream>>>(ei + N_EDGES, dinv);
    dinv_kernel<<<(N_NODES + 255) / 256, 256, 0, stream>>>(dinv);
    norm_kernel<<<(N_EDGES + 255) / 256, 256, 0, stream>>>(ei, dinv, norm);

    // 2. weight packs + MLP via WMMA
    pack_w1_kernel<<<(HIDDEN * KPAD1 + 255) / 256, 256, 0, stream>>>(W1, W1T);
    pack_w2_kernel<<<(NPAD2 * HIDDEN + 255) / 256, 256, 0, stream>>>(W2, W2T);
    gemm1_relu_kernel<<<N_NODES / 16, 256, 0, stream>>>(x, W1T, b1, h1);
    gemm2_kernel<<<N_NODES / 16, 96, 0, stream>>>(h1, W2T, b2, h0);

    // 3. APPNP propagation, K steps; last step writes d_out directly
    const int init_blocks = (N_NODES * N_CLASSES + 255) / 256;
    const int edge_blocks = ((int)((unsigned)N_EDGES * 10u) + 255) / 256;
    float* buf[2] = {hA, hB};
    for (int k = 0; k < K_STEPS; ++k) {
        const float* cur = (k == 0) ? h0 : buf[(k - 1) & 1];
        float* next = (k == K_STEPS - 1) ? out : buf[k & 1];
        prop_init_kernel<<<init_blocks, 256, 0, stream>>>(h0, cur, dinv, next);
        prop_edge_kernel<<<edge_blocks, 256, 0, stream>>>(ei, norm, cur, next);
    }
}
